// IRNetGBaBlock_87479893885462
// MI455X (gfx1250) — 8-bit matrix operand order repaired and checked by reading, compiled, not run on hardware
//
#include <hip/hip_runtime.h>
#include <hip/hip_bf16.h>
#include <math.h>

typedef int v8i __attribute__((ext_vector_type(8)));

constexpr int B_  = 64;
constexpr int C_  = 256;
constexpr int H_  = 28;
constexpr int W_  = 28;
constexpr int HW_ = H_ * W_;
constexpr int G_  = 4;
constexpr int CG_ = C_ / G_;
constexpr int K1TOT = 9 * CG_;
constexpr int K2TOT = 9 * C_;
constexpr int NT   = HW_ / 16;

constexpr size_t OFF_XMAX  = 0;
constexpr size_t OFF_W1P   = 256;
constexpr size_t OFF_W2P   = OFF_W1P + (size_t)G_ * C_ * K1TOT;
constexpr size_t OFF_S1    = OFF_W2P + (size_t)C_ * K2TOT;
constexpr size_t OFF_B1    = OFF_S1 + (size_t)G_ * C_ * sizeof(float);
constexpr size_t OFF_S2    = OFF_B1 + (size_t)G_ * C_ * sizeof(float);
constexpr size_t OFF_B2    = OFF_S2 + (size_t)C_ * sizeof(float);
constexpr size_t OFF_ABITS = OFF_B2 + (size_t)C_ * sizeof(float);
constexpr size_t OFF_OUT1  = OFF_ABITS + (size_t)B_ * HW_ * C_;
constexpr size_t OFF_AB2   = OFF_OUT1 + (size_t)B_ * C_ * HW_ * sizeof(float);

__global__ void k_init(unsigned* xmax) {
    if (blockIdx.x == 0 && threadIdx.x == 0) *xmax = 0u;
}

__global__ void k_absmax(const float* __restrict__ x, unsigned* __restrict__ xmax, int n) {
    __shared__ float red[256];
    float m = 0.f;
    for (int i = blockIdx.x * blockDim.x + threadIdx.x; i < n; i += gridDim.x * blockDim.x)
        m = fmaxf(m, fabsf(x[i]));
    red[threadIdx.x] = m;
    __syncthreads();
    for (int s = 128; s > 0; s >>= 1) {
        if (threadIdx.x < s) red[threadIdx.x] = fmaxf(red[threadIdx.x], red[threadIdx.x + s]);
        __syncthreads();
    }
    if (threadIdx.x == 0) atomicMax(xmax, __float_as_uint(red[0]));
}

__global__ void k_wprep(const float* __restrict__ w1, const float* __restrict__ g1,
                        const float* __restrict__ b1, const float* __restrict__ m1,
                        const float* __restrict__ v1,
                        const float* __restrict__ w2, const float* __restrict__ g2,
                        const float* __restrict__ b2, const float* __restrict__ m2,
                        const float* __restrict__ v2,
                        signed char* __restrict__ w1p, signed char* __restrict__ w2p,
                        float* __restrict__ s1eff, float* __restrict__ b1eff,
                        float* __restrict__ s2eff, float* __restrict__ b2eff) {
    __shared__ float rs[256], rq[256];
    __shared__ float sh_mean, sh_istd;
    const int bid = blockIdx.x;
    const float* w; signed char* pk; int n, ci;
    float gg, bb, mm, vv; float* sdst; float* bdst;
    if (bid < G_ * C_) {
        w = w1 + (size_t)bid * K1TOT;  pk = w1p + (size_t)bid * K1TOT;
        n = K1TOT; ci = CG_;
        gg = g1[bid]; bb = b1[bid]; mm = m1[bid]; vv = v1[bid];
        sdst = s1eff + bid; bdst = b1eff + bid;
    } else {
        int oc = bid - G_ * C_;
        w = w2 + (size_t)oc * K2TOT;   pk = w2p + (size_t)oc * K2TOT;
        n = K2TOT; ci = C_;
        gg = g2[oc]; bb = b2[oc]; mm = m2[oc]; vv = v2[oc];
        sdst = s2eff + oc; bdst = b2eff + oc;
    }
    const int t = threadIdx.x;
    float s = 0.f, q = 0.f;
    for (int idx = t; idx < n; idx += 256) { float vw = w[idx]; s += vw; q += vw * vw; }
    rs[t] = s; rq[t] = q; __syncthreads();
    for (int st = 128; st > 0; st >>= 1) {
        if (t < st) { rs[t] += rs[t + st]; rq[t] += rq[t + st]; }
        __syncthreads();
    }
    if (t == 0) {
        float mean = rs[0] / (float)n;
        float var  = (rq[0] - (float)n * mean * mean) / (float)(n - 1);
        sh_mean = mean; sh_istd = rsqrtf(var);
    }
    __syncthreads();
    const float mean = sh_mean, istd = sh_istd;
    float asum = 0.f;
    for (int idx = t; idx < n; idx += 256) {
        float d = w[idx] - mean;
        asum += fabsf(d) * istd;
        int ic = idx / 9, tap = idx % 9;
        pk[tap * ci + ic] = d > 0.f ? 1 : (d < 0.f ? -1 : 0);
    }
    rs[t] = asum; __syncthreads();
    for (int st = 128; st > 0; st >>= 1) {
        if (t < st) rs[t] += rs[t + st];
        __syncthreads();
    }
    if (t == 0) {
        float meanabs = rs[0] / (float)n;
        float scale   = exp2f(rintf(log2f(meanabs)));
        float inv     = gg * rsqrtf(vv + 1e-5f);
        *sdst = scale * inv;
        *bdst = bb - mm * inv;
    }
}

__global__ void k_binx(const float* __restrict__ x, const unsigned* __restrict__ xmaxb,
                       unsigned* __restrict__ abits) {
    const int tid = blockIdx.x * blockDim.x + threadIdx.x;
    const int total = B_ * HW_ * (C_ / 4);
    if (tid >= total) return;
    const int c4  = tid & 63;
    const int pos = tid >> 6;
    const int b = pos / HW_, hw = pos % HW_;
    const float xm = __uint_as_float(*xmaxb);
    const float alpha[4] = {-0.6f, -0.2f, 0.2f, 0.6f};
    unsigned outw = 0u;
    for (int k = 0; k < 4; ++k) {
        float xv = x[((size_t)(b * C_ + c4 * 4 + k)) * HW_ + hw];
        unsigned bits = 0u;
        #pragma unroll
        for (int i = 0; i < 4; ++i) bits |= (xv + alpha[i] * xm > 0.f ? 1u : 0u) << i;
        outw |= bits << (8 * k);
    }
    abits[tid] = outw;
}

__global__ __launch_bounds__(512) void k_conv1(
        const signed char* __restrict__ w1p, const unsigned* __restrict__ abits,
        const float* __restrict__ s1eff, const float* __restrict__ b1eff,
        const float* __restrict__ x,
        float* __restrict__ out1, signed char* __restrict__ abin2) {
    __shared__ int ldsA[4 * 16 * 64];
    const int tid  = threadIdx.x;
    const int lane = tid & 31, wv = tid >> 5;
    const int img   = blockIdx.x / NT;
    const int tbase = (blockIdx.x % NT) * 16;
    const int nn = lane & 15, hi = lane >> 4;
    const int grp = wv >> 2;
    const int ocbase = wv * 16;

    const v8i vzero = {0, 0, 0, 0, 0, 0, 0, 0};
    v8i acc[4] = {vzero, vzero, vzero, vzero};

    __builtin_prefetch(w1p + (size_t)ocbase * K1TOT, 0, 1);

    for (int tap = 0; tap < 9; ++tap) {
        __syncthreads();
        const int dy = tap / 3 - 1, dx = tap % 3 - 1;
        #pragma unroll
        for (int j = 0; j < 2; ++j) {
            int e  = tid + j * 512;
            int m  = e >> 6, c4 = e & 63;
            int hw = tbase + m;
            int h  = hw / W_ + dy, w = hw % W_ + dx;
            bool valid = (h >= 0 && h < H_ && w >= 0 && w < W_);
            unsigned bits = valid ? abits[(size_t)(img * HW_ + h * W_ + w) * 64 + c4] : 0u;
            #pragma unroll
            for (int br = 0; br < 4; ++br) {
                unsigned tb = (bits >> br) & 0x01010101u;
                ldsA[((br * 16 + m) << 6) + c4] = valid ? (int)~(tb * 0xFEu) : 0;
            }
        }
        __syncthreads();
        #pragma unroll
        for (int br = 0; br < 4; ++br) {
            v8i a, bf;
            #pragma unroll
            for (int v = 0; v < 8; ++v) {
                int kbase = ((v >> 1) << 4) + ((v & 1) << 2) + hi * 8;
                a[v] = ldsA[((br * 16 + nn) << 6) + grp * 16 + (kbase >> 2)];
            }
            const int* wp = (const int*)w1p +
                (((size_t)(br * C_ + ocbase + nn) * K1TOT + (size_t)tap * CG_) >> 2);
            #pragma unroll
            for (int v = 0; v < 8; ++v) {
                int kk = ((v >> 1) << 4) + ((v & 1) << 2) + hi * 8;
                bf[v] = wp[kk >> 2];
            }
            acc[br] = __builtin_amdgcn_wmma_i32_16x16x64_iu8(true, a, true, bf, acc[br],
                                                             false, false);
        }
    }

    const int ch = ocbase + nn;
    const float s0 = s1eff[0 * C_ + ch], s1 = s1eff[1 * C_ + ch];
    const float s2 = s1eff[2 * C_ + ch], s3 = s1eff[3 * C_ + ch];
    const float bsum = b1eff[0 * C_ + ch] + b1eff[1 * C_ + ch] +
                       b1eff[2 * C_ + ch] + b1eff[3 * C_ + ch];
    const size_t rowoff = (size_t)(img * C_ + ch) * HW_ + tbase + hi * 8;
    const float* xr = x + rowoff;
    float* o1 = out1 + rowoff;
    #pragma unroll
    for (int r = 0; r < 8; ++r) {
        float val = (float)acc[0][r] * s0 + (float)acc[1][r] * s1 +
                    (float)acc[2][r] * s2 + (float)acc[3][r] * s3 + bsum + xr[r];
        val = fminf(1.f, fmaxf(-1.f, val));
        o1[r] = val;
        abin2[(size_t)(img * HW_ + tbase + hi * 8 + r) * C_ + ch] =
            val > 0.f ? 1 : (val < 0.f ? -1 : 0);
    }
}

__global__ __launch_bounds__(512) void k_conv2(
        const signed char* __restrict__ w2p, const signed char* __restrict__ abin2,
        const float* __restrict__ s2eff, const float* __restrict__ b2eff,
        const float* __restrict__ out1, float* __restrict__ dout) {
    __shared__ int ldsA[16 * 64];
    const int tid  = threadIdx.x;
    const int lane = tid & 31, wv = tid >> 5;
    const int img   = blockIdx.x / NT;
    const int tbase = (blockIdx.x % NT) * 16;
    const int nn = lane & 15, hi = lane >> 4;
    const int ocbase = wv * 16;
    v8i acc = {0, 0, 0, 0, 0, 0, 0, 0};
    const int* ab = (const int*)abin2;

    __builtin_prefetch(w2p + (size_t)ocbase * K2TOT, 0, 1);

    for (int tap = 0; tap < 9; ++tap) {
        __syncthreads();
        const int dy = tap / 3 - 1, dx = tap % 3 - 1;
        #pragma unroll
        for (int j = 0; j < 2; ++j) {
            int e  = tid + j * 512;
            int m  = e >> 6, c4 = e & 63;
            int hw = tbase + m;
            int h  = hw / W_ + dy, w = hw % W_ + dx;
            ldsA[(m << 6) + c4] = (h >= 0 && h < H_ && w >= 0 && w < W_)
                ? ab[(size_t)(img * HW_ + h * W_ + w) * 64 + c4] : 0;
        }
        __syncthreads();
        #pragma unroll
        for (int cb = 0; cb < 4; ++cb) {
            v8i a, bf;
            #pragma unroll
            for (int v = 0; v < 8; ++v) {
                int kbase = ((v >> 1) << 4) + ((v & 1) << 2) + hi * 8;
                a[v] = ldsA[(nn << 6) + cb * 16 + (kbase >> 2)];
            }
            const int* wp = (const int*)w2p +
                (((size_t)(ocbase + nn) * K2TOT + (size_t)tap * C_ + (size_t)cb * CG_) >> 2);
            #pragma unroll
            for (int v = 0; v < 8; ++v) {
                int kk = ((v >> 1) << 4) + ((v & 1) << 2) + hi * 8;
                bf[v] = wp[kk >> 2];
            }
            acc = __builtin_amdgcn_wmma_i32_16x16x64_iu8(true, a, true, bf, acc, false, false);
        }
    }

    const int ch = ocbase + nn;
    const float sc = s2eff[ch], be = b2eff[ch];
    const size_t rowoff = (size_t)(img * C_ + ch) * HW_ + tbase + hi * 8;
    #pragma unroll
    for (int r = 0; r < 8; ++r) {
        float val = (float)acc[r] * sc + be + out1[rowoff + r];
        dout[rowoff + r] = fminf(1.f, fmaxf(-1.f, val));
    }
}

extern "C" void kernel_launch(void* const* d_in, const int* in_sizes, int n_in,
                              void* d_out, int out_size, void* d_ws, size_t ws_size,
                              hipStream_t stream) {
    (void)in_sizes; (void)n_in; (void)out_size; (void)ws_size;
    const float* x  = (const float*)d_in[0];
    const float* w1 = (const float*)d_in[1];
    const float* g1 = (const float*)d_in[2];
    const float* b1 = (const float*)d_in[3];
    const float* m1 = (const float*)d_in[4];
    const float* v1 = (const float*)d_in[5];
    const float* w2 = (const float*)d_in[6];
    const float* g2 = (const float*)d_in[7];
    const float* b2 = (const float*)d_in[8];
    const float* m2 = (const float*)d_in[9];
    const float* v2 = (const float*)d_in[10];

    char* ws = (char*)d_ws;
    unsigned*    xmax  = (unsigned*)(ws + OFF_XMAX);
    signed char* w1p   = (signed char*)(ws + OFF_W1P);
    signed char* w2p   = (signed char*)(ws + OFF_W2P);
    float*       s1e   = (float*)(ws + OFF_S1);
    float*       b1e   = (float*)(ws + OFF_B1);
    float*       s2e   = (float*)(ws + OFF_S2);
    float*       b2e   = (float*)(ws + OFF_B2);
    unsigned*    abits = (unsigned*)(ws + OFF_ABITS);
    float*       out1  = (float*)(ws + OFF_OUT1);
    signed char* ab2   = (signed char*)(ws + OFF_AB2);
    float*       out   = (float*)d_out;

    const int nX = B_ * C_ * HW_;

    k_init<<<1, 1, 0, stream>>>(xmax);
    k_absmax<<<1024, 256, 0, stream>>>(x, xmax, nX);
    k_wprep<<<G_ * C_ + C_, 256, 0, stream>>>(w1, g1, b1, m1, v1, w2, g2, b2, m2, v2,
                                              w1p, w2p, s1e, b1e, s2e, b2e);
    k_binx<<<(B_ * HW_ * (C_ / 4) + 255) / 256, 256, 0, stream>>>(x, xmax, abits);
    k_conv1<<<B_ * NT, 512, 0, stream>>>(w1p, abits, s1e, b1e, x, out1, ab2);
    k_conv2<<<B_ * NT, 512, 0, stream>>>(w2p, ab2, s2e, b2e, out1, out);
}
